// MultiheadIsotropicAFA_63299228008885
// MI455X (gfx1250) — matrix operand order repaired and checked by reading, not compiled, not run on hardware
//
#include <hip/hip_runtime.h>
#include <hip/hip_bf16.h>
#include <stdint.h>

// dims (match reference)
#define L_SEQ 1536
#define DE_DIM 512
#define NH 8
#define DH 32
#define EPSV 1e-6f

typedef float v8f __attribute__((ext_vector_type(8)));
typedef __bf16 v16bf __attribute__((ext_vector_type(16)));
typedef uint32_t u32x4 __attribute__((ext_vector_type(4)));
typedef float f32x4 __attribute__((ext_vector_type(4)));

struct BFPair { u32x4 a, b; };

// ---- bf16 helpers (RNE via bit math) ----
__device__ __forceinline__ uint16_t f2u16(float f) {
    uint32_t u = __builtin_bit_cast(uint32_t, f);
    u += 0x7FFFu + ((u >> 16) & 1u);
    return (uint16_t)(u >> 16);
}

__device__ __forceinline__ v16bf make_frag(u32x4 a, u32x4 b) {
    BFPair p{a, b};
    return __builtin_bit_cast(v16bf, p);
}
__device__ __forceinline__ v16bf zero_frag() {
    u32x4 z = {0u, 0u, 0u, 0u};
    return make_frag(z, z);
}

// sum of squares of 8 bf16 packed in a u32x4
__device__ __forceinline__ float sumsq8(u32x4 d) {
    float s = 0.f;
#pragma unroll
    for (int w = 0; w < 4; ++w) {
        float f0 = __builtin_bit_cast(float, d[w] << 16);
        float f1 = __builtin_bit_cast(float, d[w] & 0xFFFF0000u);
        s += f0 * f0 + f1 * f1;
    }
    return s;
}

// A-frag (16x32, lane&15 = row M): elements 0..7 <-> K = (hi?8:0)+e,
//                                  elements 8..15 <-> K = 16+(hi?8:0)+(e-8)
__device__ __forceinline__ v16bf ldA_bf16(const uint16_t* rowp, int hi) {
    const int o0 = hi ? 8 : 0;
    u32x4 lo = *(const u32x4*)(rowp + o0);
    u32x4 hi4 = *(const u32x4*)(rowp + 16 + o0);
    return make_frag(lo, hi4);
}
// B-frag (32x16, lane&15 = col N): elements 0..7 <-> K = (hi?8:0)+e,
//                                  elements 8..15 <-> K = 16+(hi?8:0)+(e-8)
__device__ __forceinline__ v16bf ldB_bf16(const uint16_t* rowp, int hi) {
    const int o0 = hi ? 8 : 0;
    u32x4 lo = *(const u32x4*)(rowp + o0);
    u32x4 hi4 = *(const u32x4*)(rowp + 16 + o0);
    return make_frag(lo, hi4);
}

// =====================================================================================
// Prep kernel A: elementwise fp32 -> bf16 (packed b128 in / b128-equivalent out)
// =====================================================================================
__global__ __launch_bounds__(256)
void cvt_bf16_kernel(const float* __restrict__ src, uint16_t* __restrict__ dst, int n8) {
    const int t = blockIdx.x * blockDim.x + threadIdx.x;
    if (t >= n8) return;
    const f32x4* s = (const f32x4*)(src + (size_t)t * 8);
    const f32x4 a = s[0], b = s[1];
    u32x4 o;
    o[0] = (uint32_t)f2u16(a[0]) | ((uint32_t)f2u16(a[1]) << 16);
    o[1] = (uint32_t)f2u16(a[2]) | ((uint32_t)f2u16(a[3]) << 16);
    o[2] = (uint32_t)f2u16(b[0]) | ((uint32_t)f2u16(b[1]) << 16);
    o[3] = (uint32_t)f2u16(b[2]) | ((uint32_t)f2u16(b[3]) << 16);
    *(u32x4*)(dst + (size_t)t * 8) = o;
}

// =====================================================================================
// Prep kernel B: transpose+convert 512x512 fp32 W[k][n] -> bf16 Wt[n][k]
// (makes GEMM B-fragments contiguous per lane). Grid (32,32), block (16,16).
// =====================================================================================
__global__ __launch_bounds__(256)
void transpose_bf16_kernel(const float* __restrict__ W, uint16_t* __restrict__ Wt) {
    __shared__ float tile[16][17];
    const int tx = threadIdx.x, ty = threadIdx.y;
    const int n0 = blockIdx.x * 16, k0 = blockIdx.y * 16;
    tile[ty][tx] = W[(size_t)(k0 + ty) * 512 + n0 + tx];
    __syncthreads();
    Wt[(size_t)(n0 + ty) * 512 + k0 + tx] = f2u16(tile[tx][ty]);
}

// =====================================================================================
// Kernel 1: projection GEMM (Zbf @ Wt^T + b), then rot_neg by position, store bf16.
// vtransposed==0: dst[(h*L + i)*64 + d] (Q/K layout)
// vtransposed==1: dst[(h*64 + d)*L + i] (V layout, feature-major => contiguous keys)
// =====================================================================================
__global__ __launch_bounds__(32)
void proj_rot_kernel(const uint16_t* __restrict__ Zbf,
                     const uint16_t* __restrict__ Wt,
                     const float* __restrict__ bvec,
                     const float* __restrict__ lambda_imag,
                     uint16_t* __restrict__ dst,
                     int vtransposed) {
    const int it = blockIdx.x, ct = blockIdx.y;
    const int lane = threadIdx.x;
    const int hi = lane >> 4, ln = lane & 15;
    const int i0 = it * 16;

    const uint16_t* arow = Zbf + (uint64_t)(i0 + ln) * DE_DIM;
    const uint16_t* brR = Wt + (uint64_t)(ct * 16 + ln) * DE_DIM;          // real cols
    const uint16_t* brI = Wt + (uint64_t)(256 + ct * 16 + ln) * DE_DIM;    // imag cols

    v8f accR = {}, accI = {};
    for (int kk = 0; kk < DE_DIM; kk += 32) {
        const v16bf a = ldA_bf16(arow + kk, hi);
        const v16bf bR = ldB_bf16(brR + kk, hi);
        const v16bf bI = ldB_bf16(brI + kk, hi);
        accR = __builtin_amdgcn_wmma_f32_16x16x32_bf16(false, a, false, bR, (short)0, accR, false, false);
        accI = __builtin_amdgcn_wmma_f32_16x16x32_bf16(false, a, false, bI, (short)0, accI, false, false);
    }

    const int c = ct * 16 + ln;         // real channel index 0..255
    const int h = c >> 5, d = c & 31;   // head, within-head dim
    const float om = lambda_imag[h * 16 + (d >> 1)];
    const float bR0 = bvec[c], bI0 = bvec[256 + c];
#pragma unroll
    for (int r = 0; r < 8; ++r) {
        const int i = i0 + r + 8 * hi;
        const float Xr = accR[r] + bR0;
        const float Xi = accI[r] + bI0;
        const float th = (float)i * om;
        const float cth = __cosf(th), sth = __sinf(th);
        const float Xr2 = Xr * cth + Xi * sth;   // rot_neg
        const float Xi2 = Xi * cth - Xr * sth;
        if (!vtransposed) {
            const uint64_t base = ((uint64_t)h * L_SEQ + i) * 64;
            dst[base + d] = f2u16(Xr2);
            dst[base + 32 + d] = f2u16(Xi2);
        } else {
            dst[((uint64_t)h * 64 + d) * L_SEQ + i] = f2u16(Xr2);
            dst[((uint64_t)h * 64 + 32 + d) * L_SEQ + i] = f2u16(Xi2);
        }
    }
}

// =====================================================================================
// Kernel 2: fused AFA attention. One wave per (16-row tile, head).
// Online softmax with register butterfly reductions; QK^T and P*V via WMMA.
// =====================================================================================
__global__ __launch_bounds__(32)
void afa_attn_kernel(const uint16_t* __restrict__ Qc,
                     const uint16_t* __restrict__ Kc,
                     const uint16_t* __restrict__ Vt,
                     const float* __restrict__ lambda_real,
                     const float* __restrict__ lambda_imag,
                     const float* __restrict__ sigma,
                     const float* __restrict__ eta,
                     const float* __restrict__ gamma_,
                     const float* __restrict__ nu_sqrt,
                     const float* __restrict__ tau,
                     uint16_t* __restrict__ est) {
    __shared__ alignas(16) uint16_t psm[16][16];  // P tile, bf16 (D-layout -> A-layout transpose)
    __shared__ float qsq[16];                     // |Q_i|^2 per row broadcast

    const int it = blockIdx.x, h = blockIdx.y;
    const int lane = threadIdx.x;
    const int hi = lane >> 4, ln = lane & 15;
    const int i0 = it * 16;

    // per-head scalars (fast-path transcendentals -> v_exp_f32 / v_log_f32)
    const float lam = -__logf(1.0f + __expf(lambda_real[h]));   // -softplus
    const float sg = sigma[h], et = eta[h], gm = gamma_[h];
    const float c0 = sg * sg / (-2.0f * lam + EPSV);
    const float eta2 = et * et, gam2 = gm * gm;
    const float nu = __logf(1.0f + __expf(nu_sqrt[h])) + EPSV;
    const float sfac = nu * (float)DH;
    const float tt = tau[h];
    const float tau2 = tt * tt;

    const uint16_t* Qh = Qc + (uint64_t)h * L_SEQ * 64;
    const uint16_t* Kh = Kc + (uint64_t)h * L_SEQ * 64;
    const uint16_t* Vh = Vt + (uint64_t)h * 64 * L_SEQ;

    // Q fragments (two K=32 A-frags over 64 rotated features) + q^2 fold
    const uint16_t* qrow = Qh + (uint64_t)(i0 + ln) * 64;
    const v16bf qa0 = ldA_bf16(qrow, hi);
    const v16bf qa1 = ldA_bf16(qrow + 32, hi);
    {
        const int o0 = hi ? 8 : 0;
        float ssq = sumsq8(*(const u32x4*)(qrow + o0)) +
                    sumsq8(*(const u32x4*)(qrow + 16 + o0)) +
                    sumsq8(*(const u32x4*)(qrow + 32 + o0)) +
                    sumsq8(*(const u32x4*)(qrow + 48 + o0));
        ssq += __shfl_xor(ssq, 16, 32);
        if (hi == 0) qsq[ln] = ssq;
    }
    __syncthreads();
    float q2r[8];
#pragma unroll
    for (int r = 0; r < 8; ++r) q2r[r] = qsq[r + 8 * hi];

    // running softmax stats per row (replicated across each 16-lane half)
    float rm_[8], rl_[8];
#pragma unroll
    for (int r = 0; r < 8; ++r) { rm_[r] = -INFINITY; rl_[r] = 0.f; }

    v8f acc0 = {}, acc1 = {}, acc2 = {}, acc3 = {};  // 16 rows x 64 features (Vtr|Vti)

    for (int jt = 0; jt <= it; ++jt) {
        const int j0 = jt * 16;

        // K B-fragments + k^2 per key
        const uint16_t* krow = Kh + (uint64_t)(j0 + ln) * 64;
        const v16bf kb0 = ldB_bf16(krow, hi);
        const v16bf kb1 = ldB_bf16(krow + 32, hi);
        float ksq;
        {
            const int o0 = hi ? 16 : 0;
            ksq = sumsq8(*(const u32x4*)(krow + o0)) +
                  sumsq8(*(const u32x4*)(krow + o0 + 8)) +
                  sumsq8(*(const u32x4*)(krow + 32 + o0)) +
                  sumsq8(*(const u32x4*)(krow + 32 + o0 + 8));
            ksq += __shfl_xor(ksq, 16, 32);
        }

        // cross[i,j] = Re(Q_rot . conj(K_rot)) over 64 features
        v8f cross = {};
        cross = __builtin_amdgcn_wmma_f32_16x16x32_bf16(false, qa0, false, kb0, (short)0, cross, false, false);
        cross = __builtin_amdgcn_wmma_f32_16x16x32_bf16(false, qa1, false, kb1, (short)0, cross, false, false);

        // scores in fp32 (D layout: lane -> key j, vgpr r -> query row)
        const int j = j0 + ln;
#pragma unroll
        for (int r = 0; r < 8; ++r) {
            const int i = i0 + r + 8 * hi;
            float sv;
            if (j > i) {
                sv = -INFINITY;
            } else {
                const float dt = (float)(i - j);
                const float dec = __expf(dt * lam);
                const float d2 = dec * dec;
                const float Vij = c0 * (1.0f - d2) + eta2 * d2 + gam2;
                float R = q2r[r] + d2 * ksq - 2.0f * dec * cross[r];
                R = fmaxf(R, 0.0f);
                sv = tau2 * (-__logf(Vij + EPSV) - __logf(1.0f + R / (sfac * Vij)));
            }
            // online softmax update via butterfly reductions within the 16-lane half
            float tmax = sv;
#pragma unroll
            for (int msk = 1; msk < 16; msk <<= 1)
                tmax = fmaxf(tmax, __shfl_xor(tmax, msk, 32));
            const float m_new = fmaxf(rm_[r], tmax);
            const float scl = __expf(rm_[r] - m_new);   // exp(-inf)=0 on first tile
            const float p = __expf(sv - m_new);
            float psum = p;
#pragma unroll
            for (int msk = 1; msk < 16; msk <<= 1)
                psum += __shfl_xor(psum, msk, 32);
            rl_[r] = rl_[r] * scl + psum;
            rm_[r] = m_new;
            acc0[r] *= scl; acc1[r] *= scl; acc2[r] *= scl; acc3[r] *= scl;
            psm[r + 8 * hi][ln] = f2u16(p);              // stage bf16 for transpose
        }
        __syncthreads();

        // P as A-frag (rows x K=32, keys 16..31 zero-padded):
        // lo half: element e <- psm[row][e] (two 16B LDS reads); hi half zero.
        const v16bf pf = hi ? zero_frag()
                            : make_frag(*(const u32x4*)(&psm[ln][0]), *(const u32x4*)(&psm[ln][8]));

        // V B-frags from transposed layout: lo half K=0..15 -> contiguous keys; hi half zero
        v16bf vb0, vb1, vb2, vb3;
        if (hi == 0) {
            const uint16_t* v0 = Vh + (uint64_t)(0 * 16 + ln) * L_SEQ + j0;
            const uint16_t* v1 = Vh + (uint64_t)(1 * 16 + ln) * L_SEQ + j0;
            const uint16_t* v2 = Vh + (uint64_t)(2 * 16 + ln) * L_SEQ + j0;
            const uint16_t* v3 = Vh + (uint64_t)(3 * 16 + ln) * L_SEQ + j0;
            vb0 = make_frag(*(const u32x4*)v0, *(const u32x4*)(v0 + 8));
            vb1 = make_frag(*(const u32x4*)v1, *(const u32x4*)(v1 + 8));
            vb2 = make_frag(*(const u32x4*)v2, *(const u32x4*)(v2 + 8));
            vb3 = make_frag(*(const u32x4*)v3, *(const u32x4*)(v3 + 8));
        } else {
            vb0 = zero_frag(); vb1 = zero_frag(); vb2 = zero_frag(); vb3 = zero_frag();
        }
        acc0 = __builtin_amdgcn_wmma_f32_16x16x32_bf16(false, pf, false, vb0, (short)0, acc0, false, false);
        acc1 = __builtin_amdgcn_wmma_f32_16x16x32_bf16(false, pf, false, vb1, (short)0, acc1, false, false);
        acc2 = __builtin_amdgcn_wmma_f32_16x16x32_bf16(false, pf, false, vb2, (short)0, acc2, false, false);
        acc3 = __builtin_amdgcn_wmma_f32_16x16x32_bf16(false, pf, false, vb3, (short)0, acc3, false, false);
        __syncthreads();  // psm reused next iteration
    }

    // epilogue: normalize, rotate to query time (+theta), store bf16 est (L x 512)
#pragma unroll
    for (int r = 0; r < 8; ++r) {
        const int i = i0 + r + 8 * hi;
        const float invl = 1.0f / rl_[r];
#pragma unroll
        for (int g = 0; g < 2; ++g) {
            const float er = (g == 0 ? acc0[r] : acc1[r]) * invl;
            const float ei = (g == 0 ? acc2[r] : acc3[r]) * invl;
            const int d = g * 16 + ln;
            const float om = lambda_imag[h * 16 + (d >> 1)];
            const float th = (float)i * om;
            const float cth = __cosf(th), sth = __sinf(th);
            const float outr = er * cth - ei * sth;
            const float outi = er * sth + ei * cth;
            est[(uint64_t)i * 512 + h * 32 + d] = f2u16(outr);
            est[(uint64_t)i * 512 + 256 + h * 32 + d] = f2u16(outi);
        }
    }
}

// =====================================================================================
// Kernel 3: output projection est(bf16, Lx512) @ Wo^T(bf16) + Wo_b -> fp32 out
// =====================================================================================
__global__ __launch_bounds__(32)
void out_proj_kernel(const uint16_t* __restrict__ est,
                     const uint16_t* __restrict__ Wot,
                     const float* __restrict__ Wob,
                     float* __restrict__ out) {
    const int it = blockIdx.x, ct = blockIdx.y;
    const int lane = threadIdx.x;
    const int hi = lane >> 4, ln = lane & 15;
    const int i0 = it * 16;

    const uint16_t* arow = est + (uint64_t)(i0 + ln) * 512;
    const uint16_t* brow = Wot + (uint64_t)(ct * 16 + ln) * 512;
    v8f acc = {};
    for (int kk = 0; kk < 512; kk += 32) {
        const v16bf a = ldA_bf16(arow + kk, hi);
        const v16bf b = ldB_bf16(brow + kk, hi);
        acc = __builtin_amdgcn_wmma_f32_16x16x32_bf16(false, a, false, b, (short)0, acc, false, false);
    }
    const float bias = Wob[ct * 16 + ln];
#pragma unroll
    for (int r = 0; r < 8; ++r) {
        const int i = i0 + r + 8 * hi;
        out[(uint64_t)i * 512 + ct * 16 + ln] = acc[r] + bias;
    }
}

// =====================================================================================
extern "C" void kernel_launch(void* const* d_in, const int* in_sizes, int n_in,
                              void* d_out, int out_size, void* d_ws, size_t ws_size,
                              hipStream_t stream) {
    (void)in_sizes; (void)n_in; (void)out_size; (void)ws_size;
    const float* Z_q = (const float*)d_in[0];
    const float* Z_k = (const float*)d_in[1];
    const float* Z_v = (const float*)d_in[2];
    const float* Wq_w = (const float*)d_in[3];
    const float* Wq_b = (const float*)d_in[4];
    const float* Wk_w = (const float*)d_in[5];
    const float* Wk_b = (const float*)d_in[6];
    const float* Wv_w = (const float*)d_in[7];
    const float* Wv_b = (const float*)d_in[8];
    const float* Wo_w = (const float*)d_in[9];
    const float* Wo_b = (const float*)d_in[10];
    const float* lambda_real = (const float*)d_in[11];
    const float* lambda_imag = (const float*)d_in[12];
    const float* sigma = (const float*)d_in[13];
    const float* eta = (const float*)d_in[14];
    const float* gamma_ = (const float*)d_in[15];
    const float* nu_sqrt = (const float*)d_in[16];
    const float* tau = (const float*)d_in[17];

    // workspace layout (uint16 units)
    const size_t HL64 = (size_t)NH * L_SEQ * 64;   // 786432
    const size_t LDE = (size_t)L_SEQ * DE_DIM;     // 786432
    const size_t WSZ = (size_t)512 * 512;          // 262144
    uint16_t* Qc = (uint16_t*)d_ws;
    uint16_t* Kc = Qc + HL64;
    uint16_t* Vt = Kc + HL64;
    uint16_t* est = Vt + HL64;
    uint16_t* Zqb = est + LDE;
    uint16_t* Zkb = Zqb + LDE;
    uint16_t* Zvb = Zkb + LDE;
    uint16_t* Wqt = Zvb + LDE;
    uint16_t* Wkt = Wqt + WSZ;
    uint16_t* Wvt = Wkt + WSZ;
    uint16_t* Wot = Wvt + WSZ;

    // --- prep: convert activations, transpose+convert weights (one-time) ---
    {
        const int n8 = (int)(LDE / 8);
        const dim3 cblk(256), cgrd((n8 + 255) / 256);
        cvt_bf16_kernel<<<cgrd, cblk, 0, stream>>>(Z_q, Zqb, n8);
        cvt_bf16_kernel<<<cgrd, cblk, 0, stream>>>(Z_k, Zkb, n8);
        cvt_bf16_kernel<<<cgrd, cblk, 0, stream>>>(Z_v, Zvb, n8);
        const dim3 tblk(16, 16), tgrd(32, 32);
        transpose_bf16_kernel<<<tgrd, tblk, 0, stream>>>(Wq_w, Wqt);
        transpose_bf16_kernel<<<tgrd, tblk, 0, stream>>>(Wk_w, Wkt);
        transpose_bf16_kernel<<<tgrd, tblk, 0, stream>>>(Wv_w, Wvt);
        transpose_bf16_kernel<<<tgrd, tblk, 0, stream>>>(Wo_w, Wot);
    }

    const dim3 blk(32);
    const dim3 gProj(L_SEQ / 16, 16);
    proj_rot_kernel<<<gProj, blk, 0, stream>>>(Zqb, Wqt, Wq_b, lambda_imag, Qc, 0);
    proj_rot_kernel<<<gProj, blk, 0, stream>>>(Zkb, Wkt, Wk_b, lambda_imag, Kc, 0);
    proj_rot_kernel<<<gProj, blk, 0, stream>>>(Zvb, Wvt, Wv_b, lambda_imag, Vt, 1);

    afa_attn_kernel<<<dim3(L_SEQ / 16, NH), blk, 0, stream>>>(
        Qc, Kc, Vt, lambda_real, lambda_imag, sigma, eta, gamma_, nu_sqrt, tau, est);

    out_proj_kernel<<<dim3(L_SEQ / 16, 32), blk, 0, stream>>>(est, Wot, Wo_b, (float*)d_out);
}
